// MultiHeadAttention_90666759618843
// MI455X (gfx1250) — matrix operand order repaired and checked by reading, not compiled, not run on hardware
//
#include <hip/hip_runtime.h>
#include <hip/hip_bf16.h>
#include <math.h>

#define D_MODEL   1024
#define NUM_HEADS 16
#define DEPTH     64
#define SEQ       2048
#define BATCH     2
#define MROWS     (BATCH * SEQ)   // 4096 rows of the [B*S, D] activation matrices
#define NCHUNK    (SEQ / 32)      // 64 key chunks per head

typedef __attribute__((ext_vector_type(16))) __bf16 v16bf;
typedef __attribute__((ext_vector_type(8)))  float  v8f;
typedef __attribute__((ext_vector_type(8)))  __bf16 bf16x8;
typedef __attribute__((ext_vector_type(4)))  int    i32x4;

// ---------------------------------------------------------------------------
// CDNA5 async-to-LDS staging (availability + signature probe-confirmed):
//   __builtin_amdgcn_global_load_async_to_lds_b128(i32x4 AS1*, i32x4 AS3*, Ii, Ii)
// ---------------------------------------------------------------------------
#if __has_builtin(__builtin_amdgcn_global_load_async_to_lds_b128)
  #define HAVE_ASYNC_LDS 1
#else
  #define HAVE_ASYNC_LDS 0
#endif

#if HAVE_ASYNC_LDS && __has_builtin(__builtin_amdgcn_s_wait_asynccnt)
  #define WAIT_ASYNC_N(n) __builtin_amdgcn_s_wait_asynccnt(n)
#elif HAVE_ASYNC_LDS
  #define WAIT_ASYNC_N(n) asm volatile("s_wait_asynccnt %0" :: "i"(n) : "memory")
#else
  #define WAIT_ASYNC_N(n) ((void)0)
#endif

// Copy 16 bytes global -> LDS. Async (ASYNCcnt) when available, else sync.
static __device__ __forceinline__ void stage_b128(__bf16* lds, const __bf16* g) {
#if HAVE_ASYNC_LDS
  // AS1 pointers are 64-bit: integer round-trip is exact.
  __attribute__((address_space(1))) i32x4* gp =
      (__attribute__((address_space(1))) i32x4*)(unsigned long long)(size_t)g;
  // AS3 pointers are 32-bit; low 32 bits of a generic LDS address are the
  // LDS offset (generic LDS addr = {SHARED_BASE[63:32], offset[31:0]}).
  __attribute__((address_space(3))) i32x4* lp =
      (__attribute__((address_space(3))) i32x4*)(unsigned int)(size_t)lds;
  __builtin_amdgcn_global_load_async_to_lds_b128(gp, lp, 0, 0);
#else
  *(uint4*)lds = *(const uint4*)g;
#endif
}

// ---------------------------------------------------------------------------
// WMMA wrapper: D = A(16x32 bf16) * B(32x16 bf16) + C(16x16 f32)
// ---------------------------------------------------------------------------
static __device__ __forceinline__ v8f wmma_bf16(v16bf a, v16bf b, v8f c) {
  return __builtin_amdgcn_wmma_f32_16x16x32_bf16(
      /*neg_a=*/false, a, /*neg_b=*/false, b,
      /*c_mod=*/(short)0, c, /*reuse_a=*/false, /*reuse_b=*/false);
}

static __device__ __forceinline__ void cvt4(v16bf& a, int base, float4 x) {
  a[base + 0] = (__bf16)x.x;
  a[base + 1] = (__bf16)x.y;
  a[base + 2] = (__bf16)x.z;
  a[base + 3] = (__bf16)x.w;
}

// ---------------------------------------------------------------------------
// Fragment builders; A and B share one contiguous k order (not AMD's table order).
// A (16x32, 16-bit): lane L holds row M = L&15.
//   lanes 0-15 : elems 0..15 -> K=k0..k0+15
//   lanes 16-31: elems 0..15 -> K=k0+16..k0+31
// B (32x16, 16-bit): lane L holds col N = L&15; lanes 0-15 K=k0+j, 16-31 K=k0+16+j
// C/D (16x16 f32): VGPR r, lane L -> (M = 8*(L>>4)+r, N = L&15)
// ---------------------------------------------------------------------------
static __device__ __forceinline__ v16bf frag_a_f32(const float* __restrict__ X,
                                                   int ld, int m0, int k0, int lane) {
  const int m  = m0 + (lane & 15);
  const int kb = k0 + ((lane >> 4) << 4);
  const float* p = X + (size_t)m * ld + kb;
  float4 x0 = *(const float4*)(p + 0);
  float4 x1 = *(const float4*)(p + 4);
  float4 x2 = *(const float4*)(p + 8);
  float4 x3 = *(const float4*)(p + 12);
  v16bf a;
  cvt4(a, 0, x0); cvt4(a, 4, x1); cvt4(a, 8, x2); cvt4(a, 12, x3);
  return a;
}

static __device__ __forceinline__ v16bf frag_a_bf16(const __bf16* __restrict__ X,
                                                    int ld, int m0, int k0, int lane) {
  const int m  = m0 + (lane & 15);
  const int kb = k0 + ((lane >> 4) << 4);
  const __bf16* p = X + (size_t)m * ld + kb;
  bf16x8 lo = *(const bf16x8*)(p);
  bf16x8 hi = *(const bf16x8*)(p + 8);
  v16bf a;
#pragma unroll
  for (int j = 0; j < 8; ++j) { a[j] = lo[j]; a[8 + j] = hi[j]; }
  return a;
}

static __device__ __forceinline__ v16bf frag_bT_f32(const float* __restrict__ W,
                                                    int ld, int n0, int k0, int lane) {
  const int n  = n0 + (lane & 15);
  const int kb = k0 + ((lane >> 4) << 4);
  const float* p = W + (size_t)n * ld + kb;
  float4 x0 = *(const float4*)(p + 0);
  float4 x1 = *(const float4*)(p + 4);
  float4 x2 = *(const float4*)(p + 8);
  float4 x3 = *(const float4*)(p + 12);
  v16bf b;
  cvt4(b, 0, x0); cvt4(b, 4, x1); cvt4(b, 8, x2); cvt4(b, 12, x3);
  return b;
}

// B^T fragment from an LDS tile [rows][64] bf16 (scores: B[k][n] = K[n][k])
static __device__ __forceinline__ v16bf frag_bT_lds(const __bf16* T,
                                                    int n0, int k0, int lane) {
  const int n  = n0 + (lane & 15);
  const int kb = k0 + ((lane >> 4) << 4);
  const __bf16* p = T + n * DEPTH + kb;
  bf16x8 lo = *(const bf16x8*)(p);
  bf16x8 hi = *(const bf16x8*)(p + 8);
  v16bf b;
#pragma unroll
  for (int j = 0; j < 8; ++j) { b[j] = lo[j]; b[8 + j] = hi[j]; }
  return b;
}

// Row-major B fragment from an LDS tile [32][64] bf16 (PV: B[k][n] = V[k][n])
static __device__ __forceinline__ v16bf frag_b_rm_lds(const __bf16* T,
                                                      int n0, int lane) {
  const int n  = n0 + (lane & 15);
  const int kb = (lane >> 4) << 4;
  v16bf b;
#pragma unroll
  for (int j = 0; j < 16; ++j) b[j] = T[(kb + j) * DEPTH + n];
  return b;
}

// ---------------------------------------------------------------------------
// Flash attention. All 8 waves of a block share one (b,h); 32-key K/V chunks
// are staged into double-buffered LDS with async-to-LDS loads pipelined one
// chunk ahead (s_wait_asynccnt 2 overlaps copy with softmax/WMMA work).
// One wave per 16-query tile.
// ---------------------------------------------------------------------------
__global__ void attn_kernel(const __bf16* __restrict__ Qh,
                            const __bf16* __restrict__ Kh,
                            const __bf16* __restrict__ Vh,
                            __bf16* __restrict__ AO) {
  __shared__ __bf16 Ks[2][32 * DEPTH];   // 2 x 4 KB: K chunk, key-major [32][64]
  __shared__ __bf16 Vs[2][32 * DEPTH];   // 2 x 4 KB: V chunk, key-major [32][64]
  __shared__ __bf16 Pl[8][16][32];       // 8 KB: per-wave P re-layout buffer

  const int lane = threadIdx.x & 31;
  const int wave = threadIdx.x >> 5;
  const int bh = blockIdx.x >> 4;                 // 32 (b*H+h), 16 blocks each
  const int qt = (blockIdx.x & 15) * 8 + wave;    // query tile in [0,128)
  const int b  = bh >> 4;
  const int h  = bh & 15;

  const __bf16* Qb = Qh + ((size_t)bh * SEQ << 6);
  const __bf16* Kb = Kh + ((size_t)bh * SEQ << 6);
  const __bf16* Vb = Vh + ((size_t)bh * SEQ << 6);

  // Q fragments for the whole tile (depth 64 = two K=32 steps)
  const v16bf qa0 = frag_a_bf16(Qb, DEPTH, qt * 16, 0,  lane);
  const v16bf qa1 = frag_a_bf16(Qb, DEPTH, qt * 16, 32, lane);

  v8f o0 = {}, o1 = {}, o2 = {}, o3 = {};
  float mrow[8], lrow[8];
#pragma unroll
  for (int r = 0; r < 8; ++r) { mrow[r] = -1e30f; lrow[r] = 0.0f; }

  const int idx = threadIdx.x * 8;   // this thread's 16B slice of the 4KB tile

  // prologue: stage chunk 0 into buffer 0
  stage_b128(Ks[0] + idx, Kb + idx);
  stage_b128(Vs[0] + idx, Vb + idx);

  for (int it = 0; it < NCHUNK; ++it) {
    const int cur  = it & 1;
    const int key0 = it * 32;

    // ---- pipeline: issue next chunk's async loads, then drain older pair ----
    if (it + 1 < NCHUNK) {
      const size_t noff = (size_t)(key0 + 32) * DEPTH + idx;
      stage_b128(Ks[cur ^ 1] + idx, Kb + noff);
      stage_b128(Vs[cur ^ 1] + idx, Vb + noff);
      WAIT_ASYNC_N(2);     // async loads retire in order: chunk `it` is home
    } else {
      WAIT_ASYNC_N(0);
    }
    __syncthreads();

    const __bf16* Kc = Ks[cur];
    const __bf16* Vc = Vs[cur];

    // ---- scores: two 16x16 tiles over 32 keys, K-dim = depth(64) ----
    v8f s0 = {}, s1 = {};
    {
      v16bf kb0 = frag_bT_lds(Kc, 0,  0,  lane);
      v16bf kb1 = frag_bT_lds(Kc, 0,  32, lane);
      s0 = wmma_bf16(qa0, kb0, s0);
      s0 = wmma_bf16(qa1, kb1, s0);
      v16bf kb2 = frag_bT_lds(Kc, 16, 0,  lane);
      v16bf kb3 = frag_bT_lds(Kc, 16, 32, lane);
      s1 = wmma_bf16(qa0, kb2, s1);
      s1 = wmma_bf16(qa1, kb3, s1);
    }

    // ---- online softmax (row stats across 16 lanes of a half-wave) ----
    float alpha[8];
#pragma unroll
    for (int r = 0; r < 8; ++r) {
      float mx = fmaxf(s0[r], s1[r]);
#pragma unroll
      for (int d = 8; d >= 1; d >>= 1) mx = fmaxf(mx, __shfl_xor(mx, d, 32));
      const float mn = fmaxf(mrow[r], mx);
      const float a  = __expf(mrow[r] - mn);
      const float p0 = __expf(s0[r] - mn);
      const float p1 = __expf(s1[r] - mn);
      float rs = p0 + p1;
#pragma unroll
      for (int d = 8; d >= 1; d >>= 1) rs += __shfl_xor(rs, d, 32);
      lrow[r]  = lrow[r] * a + rs;
      mrow[r]  = mn;
      alpha[r] = a;
      s0[r] = p0; s1[r] = p1;
    }
#pragma unroll
    for (int r = 0; r < 8; ++r) {
      o0[r] *= alpha[r]; o1[r] *= alpha[r];
      o2[r] *= alpha[r]; o3[r] *= alpha[r];
    }

    // ---- P: accumulator layout -> A-fragment layout via LDS ----
    const int mb = (lane >> 4) << 3;
#pragma unroll
    for (int r = 0; r < 8; ++r) {
      Pl[wave][mb + r][lane & 15]        = (__bf16)s0[r];
      Pl[wave][mb + r][16 + (lane & 15)] = (__bf16)s1[r];
    }
    __syncthreads();
    v16bf pa;
    {
      const int m = lane & 15;
      const int kbase = (lane >> 4) << 4;
#pragma unroll
      for (int j = 0; j < 8; ++j) {
        pa[j]     = Pl[wave][m][kbase + j];
        pa[8 + j] = Pl[wave][m][kbase + 8 + j];
      }
    }

    // ---- V fragments from LDS, then O += P @ V ----
    v16bf vb0 = frag_b_rm_lds(Vc, 0,  lane);
    v16bf vb1 = frag_b_rm_lds(Vc, 16, lane);
    v16bf vb2 = frag_b_rm_lds(Vc, 32, lane);
    v16bf vb3 = frag_b_rm_lds(Vc, 48, lane);
    __syncthreads();   // all reads of buf `cur` done before it is re-staged

    o0 = wmma_bf16(pa, vb0, o0);
    o1 = wmma_bf16(pa, vb1, o1);
    o2 = wmma_bf16(pa, vb2, o2);
    o3 = wmma_bf16(pa, vb3, o3);
  }

  // ---- finalize: O /= l, write merged-head layout [B*S, 1024] bf16 ----
  const int mb = (lane >> 4) << 3;
#pragma unroll
  for (int r = 0; r < 8; ++r) {
    const float inv = 1.0f / lrow[r];
    const int m = b * SEQ + qt * 16 + mb + r;
    const size_t base = (size_t)m * D_MODEL + h * 64 + (lane & 15);
    AO[base + 0]  = (__bf16)(o0[r] * inv);
    AO[base + 16] = (__bf16)(o1[r] * inv);
    AO[base + 32] = (__bf16)(o2[r] * inv);
    AO[base + 48] = (__bf16)(o3[r] * inv);
  }
}

// ---------------------------------------------------------------------------
// Head projection  Yh = split_heads((X @ W^T + bias) * scale)
// ---------------------------------------------------------------------------
__global__ void proj_head_kernel(const float* __restrict__ X,
                                 const float* __restrict__ W,
                                 const float* __restrict__ bias,
                                 __bf16* __restrict__ outHead,
                                 float scale) {
  const int lane = threadIdx.x & 31;
  const int wave = threadIdx.x >> 5;
  const int m0 = blockIdx.x * 16;
  const int n0 = blockIdx.y * 256 + wave * 32;

  v8f acc[2] = {v8f{}, v8f{}};
  for (int k0 = 0; k0 < D_MODEL; k0 += 32) {
    if (k0 + 32 < D_MODEL) {
      __builtin_prefetch(W + (size_t)(n0 + (lane & 15)) * D_MODEL + k0 + 32, 0, 1);
    }
    v16bf a  = frag_a_f32(X, D_MODEL, m0, k0, lane);
    v16bf b0 = frag_bT_f32(W, D_MODEL, n0,      k0, lane);
    v16bf b1 = frag_bT_f32(W, D_MODEL, n0 + 16, k0, lane);
    acc[0] = wmma_bf16(a, b0, acc[0]);
    acc[1] = wmma_bf16(a, b1, acc[1]);
  }

  const int mBase = m0 + ((lane >> 4) << 3);
#pragma unroll
  for (int t = 0; t < 2; ++t) {
    const int n = n0 + t * 16 + (lane & 15);
    const float bv = bias[n];
    const int h = n >> 6, d = n & 63;
#pragma unroll
    for (int r = 0; r < 8; ++r) {
      const int m = mBase + r;
      const int bb = m >> 11;
      const int s  = m & (SEQ - 1);
      const float y = (acc[t][r] + bv) * scale;
      outHead[(((size_t)(bb * NUM_HEADS + h) * SEQ + s) << 6) + d] = (__bf16)y;
    }
  }
}

// ---------------------------------------------------------------------------
// Output projection  Y = AO @ wo^T + bo   (f32 output)
// ---------------------------------------------------------------------------
__global__ void outproj_kernel(const __bf16* __restrict__ X,
                               const float* __restrict__ W,
                               const float* __restrict__ bias,
                               float* __restrict__ Y) {
  const int lane = threadIdx.x & 31;
  const int wave = threadIdx.x >> 5;
  const int m0 = blockIdx.x * 16;
  const int n0 = blockIdx.y * 256 + wave * 32;

  v8f acc[2] = {v8f{}, v8f{}};
  for (int k0 = 0; k0 < D_MODEL; k0 += 32) {
    if (k0 + 32 < D_MODEL) {
      __builtin_prefetch(W + (size_t)(n0 + (lane & 15)) * D_MODEL + k0 + 32, 0, 1);
    }
    v16bf a  = frag_a_bf16(X, D_MODEL, m0, k0, lane);
    v16bf b0 = frag_bT_f32(W, D_MODEL, n0,      k0, lane);
    v16bf b1 = frag_bT_f32(W, D_MODEL, n0 + 16, k0, lane);
    acc[0] = wmma_bf16(a, b0, acc[0]);
    acc[1] = wmma_bf16(a, b1, acc[1]);
  }

  const int mBase = m0 + ((lane >> 4) << 3);
#pragma unroll
  for (int t = 0; t < 2; ++t) {
    const int n = n0 + t * 16 + (lane & 15);
    const float bv = bias[n];
#pragma unroll
    for (int r = 0; r < 8; ++r) {
      const int m = mBase + r;
      Y[(size_t)m * D_MODEL + n] = acc[t][r] + bv;
    }
  }
}

// ---------------------------------------------------------------------------
extern "C" void kernel_launch(void* const* d_in, const int* in_sizes, int n_in,
                              void* d_out, int out_size, void* d_ws, size_t ws_size,
                              hipStream_t stream) {
  (void)in_sizes; (void)n_in; (void)out_size; (void)ws_size;

  const float* q  = (const float*)d_in[0];
  const float* k  = (const float*)d_in[1];
  const float* v  = (const float*)d_in[2];
  const float* wq = (const float*)d_in[3];
  const float* bq = (const float*)d_in[4];
  const float* wk = (const float*)d_in[5];
  const float* bk = (const float*)d_in[6];
  const float* wv = (const float*)d_in[7];
  const float* bv = (const float*)d_in[8];
  const float* wo = (const float*)d_in[9];
  const float* bo = (const float*)d_in[10];
  float* out = (float*)d_out;

  char* ws = (char*)d_ws;
  const size_t headBytes = (size_t)BATCH * NUM_HEADS * SEQ * DEPTH * sizeof(__bf16); // 8 MB
  __bf16* Qh = (__bf16*)(ws + 0 * headBytes);
  __bf16* Kh = (__bf16*)(ws + 1 * headBytes);
  __bf16* Vh = (__bf16*)(ws + 2 * headBytes);
  __bf16* AO = (__bf16*)(ws + 3 * headBytes);   // [MROWS, 1024] bf16, 8 MB

  const dim3 blk(256, 1, 1);
  const dim3 gProj(MROWS / 16, D_MODEL / 256, 1);   // (256, 4)
  const float qscale = 0.125f;                       // 1/sqrt(depth)

  proj_head_kernel<<<gProj, blk, 0, stream>>>(q, wq, bq, Qh, qscale);
  proj_head_kernel<<<gProj, blk, 0, stream>>>(k, wk, bk, Kh, 1.0f);
  proj_head_kernel<<<gProj, blk, 0, stream>>>(v, wv, bv, Vh, 1.0f);

  const int nblocks = BATCH * NUM_HEADS * 16;        // 512 blocks, 8 waves each
  attn_kernel<<<dim3(nblocks, 1, 1), blk, 0, stream>>>(Qh, Kh, Vh, AO);

  outproj_kernel<<<gProj, blk, 0, stream>>>(AO, wo, bo, out);
}
